// WindowExdAttention_29850022707950
// MI455X (gfx1250) — matrix operand order repaired and checked by reading, not compiled, not run on hardware
//
#include <hip/hip_runtime.h>

typedef __attribute__((ext_vector_type(4))) unsigned int u32x4;
typedef __attribute__((ext_vector_type(8))) unsigned int v8u;
typedef __attribute__((ext_vector_type(8))) float       v8f;
typedef __attribute__((ext_vector_type(16))) __bf16     v16bf;
typedef __attribute__((ext_vector_type(2))) __bf16      v2bf;
typedef __attribute__((ext_vector_type(8))) int         v8i;
typedef __attribute__((ext_vector_type(4))) int         v4i;

#define BB    16
#define CC    112
#define HWN   4096
#define NHEAD 4
#define HDIM  28
#define TOPKN 128
#define QK_SCALE 0.1889822365046136f  /* 28^-0.5 */

// TDM path only where the builtin exists on the device pass
#if defined(__HIP_DEVICE_COMPILE__) && __has_builtin(__builtin_amdgcn_tensor_load_to_lds)
#define USE_TDM 1
#if __has_include(<hip/amd_detail/amd_gfx1250_TDM.h>)
#define TDM_ARITY6 1     // therock/clang-23 headers -> 6-arg builtin
#else
#define TDM_ARITY6 0     // ROCm 7.2 / clang-22 -> 5-arg builtin
#endif
#else
#define USE_TDM 0
#endif

// ---- f32 -> bf16 via fptrunc: lets ISel pick a native cvt if gfx1250 has one ----
__device__ __forceinline__ unsigned short f2bf(float a) {
  __bf16 t = (__bf16)a;
  return __builtin_bit_cast(unsigned short, t);
}
__device__ __forceinline__ unsigned int packbf(float a, float b) {
  v2bf r; r[0] = (__bf16)a; r[1] = (__bf16)b;
  return __builtin_bit_cast(unsigned int, r);
}

__device__ __forceinline__ v8f wmma_bf16(v8u a, v8u b, v8f c) {
  return __builtin_amdgcn_wmma_f32_16x16x32_bf16(
      false, __builtin_bit_cast(v16bf, a),
      false, __builtin_bit_cast(v16bf, b),
      (short)0, c, false, false);
}
// Two 16B-aligned 8x-bf16 halves -> one 8-VGPR WMMA operand
__device__ __forceinline__ v8u load8(const unsigned short* p0, const unsigned short* p1) {
  u32x4 a = *(const u32x4*)p0;
  u32x4 b = *(const u32x4*)p1;
  v8u r;
  r[0] = a[0]; r[1] = a[1]; r[2] = a[2]; r[3] = a[3];
  r[4] = b[0]; r[5] = b[1]; r[6] = b[2]; r[7] = b[3];
  return r;
}

__global__ void zero_kernel(u32x4* __restrict__ p, long n) {
  long i = (long)blockIdx.x * blockDim.x + threadIdx.x;
  long stride = (long)gridDim.x * blockDim.x;
  u32x4 z = {0u, 0u, 0u, 0u};
  for (; i < n; i += stride) p[i] = z;
}

// Pre-swizzle weights into B-operand lane layout (zero-padded for k >= 112)
#define WQ_ELEMS (21 * 4 * 32 * 16)
#define WP_ELEMS (7 * 4 * 32 * 16)
__global__ void wprep_kernel(const float* __restrict__ Wqkv, const float* __restrict__ Wproj,
                             unsigned short* __restrict__ wqb, unsigned short* __restrict__ wpb) {
  int i = blockIdx.x * blockDim.x + threadIdx.x;
  if (i >= WQ_ELEMS + WP_ELEMS) return;
  bool isq = i < WQ_ELEMS;
  int base = isq ? i : (i - WQ_ELEMS);
  int e    = base & 15;
  int lane = (base >> 4) & 31;
  int tile = base >> 9;                       // nt*4 + kc
  int kc = tile & 3, nt = tile >> 2;
  int k = kc * 32 + ((e < 8) ? (lane >> 4) * 8 + e : 16 + (lane >> 4) * 8 + (e - 8));
  int d = nt * 16 + (lane & 15);
  const float* W = isq ? Wqkv : Wproj;
  float v = (k < CC) ? W[(size_t)d * CC + k] : 0.f;
  (isq ? wqb : wpb)[base] = f2bf(v);
}

// qkv = xT @ Wqkv^T ; q scaled; stored bf16 (B, HW, head*32+hd) padded rows
__global__ __launch_bounds__(32) void qkv_kernel(
    const float* __restrict__ x, const unsigned short* __restrict__ wqb,
    unsigned short* __restrict__ qb, unsigned short* __restrict__ kb,
    unsigned short* __restrict__ vb) {
  const int lane = threadIdx.x;
  const int h = lane >> 4, ln = lane & 15;
  const int hw0 = blockIdx.x * 16;
  const int b = blockIdx.y;

  // A tile: 16(hw) x 128(c padded), A-layout (ISA 7.12.2); x is channels-first
  v8u A[4];
#pragma unroll
  for (int kc = 0; kc < 4; ++kc) {
    v8u a;
#pragma unroll
    for (int j = 0; j < 8; ++j) {
      int k = h * 8 + ((j < 4) ? 2 * j : 8 + 2 * j);
      int c = kc * 32 + k;
      float f0 = (c < CC)     ? x[(size_t)(b * CC + c)     * HWN + hw0 + ln] : 0.f;
      float f1 = (c + 1 < CC) ? x[(size_t)(b * CC + c + 1) * HWN + hw0 + ln] : 0.f;
      a[j] = packbf(f0, f1);
    }
    A[kc] = a;
  }

  const v8u* wqv = (const v8u*)wqb;
#pragma unroll 1
  for (int section = 0; section < 3; ++section) {   // 0=q 1=k 2=v
    unsigned short* dst = (section == 0) ? qb : ((section == 1) ? kb : vb);
    float scale = (section == 0) ? QK_SCALE : 1.f;
#pragma unroll
    for (int j7 = 0; j7 < 7; ++j7) {                // head/slot math folds per tile
      int nt = section * 7 + j7;
      v8f acc = {0.f, 0.f, 0.f, 0.f, 0.f, 0.f, 0.f, 0.f};
#pragma unroll
      for (int kc = 0; kc < 4; ++kc)                // pure b128 loads + WMMA
        acc = wmma_bf16(A[kc], wqv[(nt * 4 + kc) * 32 + lane], acc);
      int cc = j7 * 16 + ln;
      int slot = (cc / HDIM) * 32 + (cc % HDIM);    // j7 constant -> cheap
#pragma unroll
      for (int r = 0; r < 8; ++r) {
        int row = hw0 + r + 8 * h;                  // C/D layout: m = r + 8*half
        dst[(size_t)(b * HWN + row) * 128 + slot] = f2bf(acc[r] * scale);
      }
    }
  }
}

// one wave per (b, window, head): TDM-gather K/V -> QK^T -> softmax -> AV
__global__ __launch_bounds__(32) void attn_kernel(
    const unsigned short* __restrict__ qb, const unsigned short* __restrict__ kb,
    const unsigned short* __restrict__ vb, const int* __restrict__ topk,
    unsigned short* __restrict__ ob) {
  __shared__ float          S[16 * 128];
  __shared__ unsigned short P[16 * 128];
#if USE_TDM
  __shared__ unsigned short Kld[TOPKN * 32];  // gathered Ksel rows (row-major)
  __shared__ unsigned short Vld[TOPKN * 32];  // gathered Vsel rows (row-major)
#else
  __shared__ unsigned short Vt[32 * 128];     // transposed Vsel: [hd][t]
#endif

  const int lane = threadIdx.x;
  const int h = lane >> 4, ln = lane & 15;
  const int id = blockIdx.x;
  const int b = id >> 10, rem = id & 1023;
  const int win = rem >> 2, head = rem & 3;
  const int wh = win >> 4, ww = win & 15;
  const int* tk = topk + win * TOPKN;

  // Q A-operand: lane row m=ln maps to window position
  int hwq = (wh * 4 + (ln >> 2)) * 64 + ww * 4 + (ln & 3);
  const unsigned short* qrow = qb + (size_t)(b * HWN + hwq) * 128 + head * 32;
  v8u Aq = load8(qrow + h * 16, qrow + h * 16 + 8);

  v8f acc[8];

#if USE_TDM
  // ---- Tensor Data Mover gather: 8 descriptors x 16 rows, for K and V ----
  // D# group1: mask=0, data_size=2B(code 1), tensor_dim0=32, tensor_dim1=4096,
  //            tile_dim0=32, tile_dim1=16 valid indices, dim0_stride=128
  v8i g1 = { (int)0x00010000, 32 << 16, 4096 << 16, 32 << 16, 16, 128, 0, 0 };
  unsigned long long gaK = (unsigned long long)(const void*)kb +
                           ((size_t)b * HWN * 128 + head * 32) * 2;
  unsigned long long gaV = (unsigned long long)(const void*)vb +
                           ((size_t)b * HWN * 128 + head * 32) * 2;
  unsigned int ldsK = (unsigned int)(unsigned long long)(const void*)&Kld[0];
  unsigned int ldsV = (unsigned int)(unsigned long long)(const void*)&Vld[0];
  const v4i* tk4 = (const v4i*)tk;            // uniform (win is wave-uniform)
#pragma unroll 1
  for (int dt = 0; dt < 8; ++dt) {
    v4i i0 = tk4[dt * 4 + 0], i1 = tk4[dt * 4 + 1];
    v4i i2 = tk4[dt * 4 + 2], i3 = tk4[dt * 4 + 3];
    v4i g2 = { (i0[0] & 0xFFFF) | (i0[1] << 16), (i0[2] & 0xFFFF) | (i0[3] << 16),
               (i1[0] & 0xFFFF) | (i1[1] << 16), (i1[2] & 0xFFFF) | (i1[3] << 16) };
    v4i g3 = { (i2[0] & 0xFFFF) | (i2[1] << 16), (i2[2] & 0xFFFF) | (i2[3] << 16),
               (i3[0] & 0xFFFF) | (i3[1] << 16), (i3[2] & 0xFFFF) | (i3[3] << 16) };
    // group0: count=1, gather_mode=1 (16-bit idx), lds_addr, global tile addr, type=2
    u32x4 g0k = { 0x80000001u, ldsK + (unsigned)dt * 1024u,
                  (unsigned int)gaK, ((unsigned int)(gaK >> 32) & 0x01FFFFFFu) | 0x80000000u };
    u32x4 g0v = { 0x80000001u, ldsV + (unsigned)dt * 1024u,
                  (unsigned int)gaV, ((unsigned int)(gaV >> 32) & 0x01FFFFFFu) | 0x80000000u };
#if TDM_ARITY6
    v8i zz = {0, 0, 0, 0, 0, 0, 0, 0};
    __builtin_amdgcn_tensor_load_to_lds(g0k, g1, g2, g3, zz, 0);
    __builtin_amdgcn_tensor_load_to_lds(g0v, g1, g2, g3, zz, 0);
#else
    __builtin_amdgcn_tensor_load_to_lds(g0k, g1, g2, g3, 0);
    __builtin_amdgcn_tensor_load_to_lds(g0v, g1, g2, g3, 0);
#endif
  }
  __builtin_amdgcn_s_wait_tensorcnt(0);
  __syncthreads();

  // S = Q(16x32) @ Ksel^T(32x128): B-operand straight from gathered LDS rows
#pragma unroll
  for (int nt = 0; nt < 8; ++nt) {
    const unsigned short* krow = &Kld[(nt * 16 + ln) * 32];
    v8u Bk = load8(krow + h * 16, krow + h * 16 + 8);
    v8f z = {0.f, 0.f, 0.f, 0.f, 0.f, 0.f, 0.f, 0.f};
    acc[nt] = wmma_bf16(Aq, Bk, z);
  }
#else
  // ---- fallback: per-lane global gather ----
#pragma unroll
  for (int nt = 0; nt < 8; ++nt) {
    int t = nt * 16 + ln;
    int idx = tk[t];
    const unsigned short* krow = kb + (size_t)(b * HWN + idx) * 128 + head * 32;
    v8u Bk = load8(krow + h * 16, krow + h * 16 + 8);
    v8f z = {0.f, 0.f, 0.f, 0.f, 0.f, 0.f, 0.f, 0.f};
    acc[nt] = wmma_bf16(Aq, Bk, z);
  }
  for (int t = lane; t < TOPKN; t += 32) {
    int idx = tk[t];
    const unsigned short* vrow = vb + (size_t)(b * HWN + idx) * 128 + head * 32;
    const u32x4* vp = (const u32x4*)vrow;
    u32x4 w0 = vp[0], w1 = vp[1], w2 = vp[2], w3 = vp[3];
    unsigned int w[16];
#pragma unroll
    for (int e = 0; e < 4; ++e) { w[e] = w0[e]; w[4 + e] = w1[e]; w[8 + e] = w2[e]; w[12 + e] = w3[e]; }
#pragma unroll
    for (int e = 0; e < 16; ++e) {
      Vt[(2 * e) * 128 + t]     = (unsigned short)(w[e] & 0xFFFFu);
      Vt[(2 * e + 1) * 128 + t] = (unsigned short)(w[e] >> 16);
    }
  }
#endif

#pragma unroll
  for (int nt = 0; nt < 8; ++nt)
#pragma unroll
    for (int r = 0; r < 8; ++r)
      S[(r + 8 * h) * 128 + nt * 16 + ln] = acc[nt][r];
  __syncthreads();

  // softmax: one row per lane (rows 0..15)
  if (lane < 16) {
    float* sr = &S[lane * 128];
    float mx = -3.0e38f;
    for (int t = 0; t < 128; ++t) mx = fmaxf(mx, sr[t]);
    float sum = 0.f;
    for (int t = 0; t < 128; ++t) { float e = __expf(sr[t] - mx); sr[t] = e; sum += e; }
    float inv = 1.f / sum;
    unsigned int* pr = (unsigned int*)&P[lane * 128];
    for (int t = 0; t < 128; t += 2)
      pr[t >> 1] = packbf(sr[t] * inv, sr[t + 1] * inv);
  }
  __syncthreads();

  // O(16x28) = P(16x128) @ Vsel(128x28): 2 N-tiles x 4 K-chunks
#pragma unroll
  for (int vt = 0; vt < 2; ++vt) {
    v8f acco = {0.f, 0.f, 0.f, 0.f, 0.f, 0.f, 0.f, 0.f};
#pragma unroll
    for (int kc = 0; kc < 4; ++kc) {
      const unsigned short* prow = &P[ln * 128 + kc * 32];
      v8u Ap = load8(prow + h * 16, prow + h * 16 + 8);
      v8u Bv;
#if USE_TDM
      int hd0 = vt * 16 + ln;
#pragma unroll
      for (int j = 0; j < 8; ++j) {                 // pair rows k,k+1 of Vsel
        int k = kc * 32 + h * 16 + 2 * j;
        Bv[j] = (unsigned int)Vld[k * 32 + hd0] | ((unsigned int)Vld[(k + 1) * 32 + hd0] << 16);
      }
#else
      const unsigned short* vr = &Vt[(vt * 16 + ln) * 128 + kc * 32];
      Bv = load8(vr + h * 16, vr + h * 16 + 8);
#endif
      acco = wmma_bf16(Ap, Bv, acco);
    }
    int hd = vt * 16 + ln;
    if (hd < HDIM) {
#pragma unroll
      for (int r = 0; r < 8; ++r) {
        int mm = r + 8 * h;
        int hw = (wh * 4 + (mm >> 2)) * 64 + ww * 4 + (mm & 3);
        ob[(size_t)(b * HWN + hw) * 128 + head * HDIM + hd] = f2bf(acco[r]);
      }
    }
  }
}

// out = o @ Wproj^T + bproj, stored transposed to (B, C, H, W)
__global__ __launch_bounds__(32) void proj_kernel(
    const unsigned short* __restrict__ ob, const unsigned short* __restrict__ wpb,
    const float* __restrict__ bproj, float* __restrict__ out) {
  const int lane = threadIdx.x;
  const int h = lane >> 4, ln = lane & 15;
  const int hw0 = blockIdx.x * 16;
  const int b = blockIdx.y;

  const unsigned short* orow = ob + (size_t)(b * HWN + hw0 + ln) * 128;
  v8u A[4];
#pragma unroll
  for (int kc = 0; kc < 4; ++kc)
    A[kc] = load8(orow + kc * 32 + h * 8, orow + kc * 32 + 16 + h * 8);

  const v8u* wpv = (const v8u*)wpb;
  for (int nt = 0; nt < 7; ++nt) {
    v8f acc = {0.f, 0.f, 0.f, 0.f, 0.f, 0.f, 0.f, 0.f};
#pragma unroll
    for (int kc = 0; kc < 4; ++kc)              // pure b128 loads + WMMA
      acc = wmma_bf16(A[kc], wpv[(nt * 4 + kc) * 32 + lane], acc);

    int d0 = nt * 16;
    float bias = bproj[d0 + ln];
    float* op = out + ((size_t)b * CC + d0 + ln) * HWN + hw0 + 8 * h;
#pragma unroll
    for (int r = 0; r < 8; ++r) op[r] = acc[r] + bias;   // 8 contiguous floats/lane
  }
}

extern "C" void kernel_launch(void* const* d_in, const int* in_sizes, int n_in,
                              void* d_out, int out_size, void* d_ws, size_t ws_size,
                              hipStream_t stream) {
  (void)in_sizes; (void)n_in; (void)out_size; (void)ws_size;
  const float* x     = (const float*)d_in[0];
  const float* Wqkv  = (const float*)d_in[1];
  const float* Wproj = (const float*)d_in[2];
  const float* bproj = (const float*)d_in[3];
  const int*   topk  = (const int*)d_in[4];
  float*       out   = (float*)d_out;

  const size_t N1 = (size_t)BB * HWN * 128;   // bf16 elems per q/k/v/o buffer
  unsigned short* qb  = (unsigned short*)d_ws;
  unsigned short* kb  = qb + N1;
  unsigned short* vb  = kb + N1;
  unsigned short* ob  = vb + N1;
  unsigned short* wqb = ob + N1;              // 32B-aligned (offset 64 MiB)
  unsigned short* wpb = wqb + WQ_ELEMS;

  long nvec = (long)(3 * N1 * sizeof(unsigned short)) / 16;
  zero_kernel<<<2048, 256, 0, stream>>>((u32x4*)qb, nvec);
  wprep_kernel<<<(WQ_ELEMS + WP_ELEMS + 255) / 256, 256, 0, stream>>>(Wqkv, Wproj, wqb, wpb);

  qkv_kernel<<<dim3(HWN / 16, BB), 32, 0, stream>>>(x, wqb, qb, kb, vb);
  attn_kernel<<<BB * 256 * NHEAD, 32, 0, stream>>>(qb, kb, vb, topk, ob);
  proj_kernel<<<dim3(HWN / 16, BB), 32, 0, stream>>>(ob, wpb, bproj, out);
}
